// GSN_matrix_8392366096418
// MI455X (gfx1250) — matrix operand order repaired and checked by reading, not compiled, not run on hardware
//
#include <hip/hip_runtime.h>
#include <hip/hip_bf16.h>
#include <math.h>

typedef __attribute__((ext_vector_type(16))) _Float16 v16h;
typedef __attribute__((ext_vector_type(8)))  float    v8f;

#define T_STEPS 32
#define BN      256
#define D_IN    128
#define H_      300
#define HP      320
#define A_      600
#define AP      608
#define G4P     1280
#define G3P     1824
#define BNT     8192
#define BNN     16384
#define CHUNK   2048

__device__ __forceinline__ float sigf(float x) { return 1.f / (1.f + __expf(-x)); }

// ---------------- zero fill (graph-capture-safe memset) ----------------
__global__ void fill_zero(float* __restrict__ p, long long n) {
  long long i = (long long)blockIdx.x * blockDim.x + threadIdx.x;
  if (i < n) p[i] = 0.f;
}

// ---------------- weight repack -> WMMA B-panel (f16) ----------------
// W: (Nout x K) f32 row-major. Panel: tile = nt*Ktiles+kt; per tile 32 lanes x 16 halfs.
// lane: N = nt*16 + (lane&15); K = kt*32 + (lane>>4)*8 + e for e<8, kt*32 + 16 + (lane>>4)*8 + (e-8) for e>=8
// padded col n -> orig col: g = n/gateSizePad, j = n%gateSizePad, oc = g*gateSize+j
// (columns that reverse-map out of range are zero-filled, so Ntiles may be padded)
__global__ void repack_w(const float* __restrict__ W, int Nout, int K,
                         int gateSize, int gateSizePad,
                         int Ntiles, int Ktiles, _Float16* __restrict__ panel)
{
  long long tid = (long long)blockIdx.x * blockDim.x + threadIdx.x;
  long long total = (long long)Ntiles * Ktiles * 32;
  if (tid >= total) return;
  int lane = (int)(tid & 31);
  long long tile = tid >> 5;
  int kt = (int)(tile % Ktiles);
  int nt = (int)(tile / Ktiles);
  int n = nt * 16 + (lane & 15);
  int g = n / gateSizePad, j = n % gateSizePad;
  int oc = g * gateSize + j;
  bool cv = (j < gateSize) && (oc < Nout);
  int kbase = kt * 32 + (lane >> 4) * 8;
  v16h o;
  #pragma unroll
  for (int e = 0; e < 16; ++e) {
    int k = kbase + ((e < 8) ? e : (16 + e - 8));
    float f = (cv && k < K) ? W[(long long)oc * K + k] : 0.f;
    o[e] = (_Float16)f;
  }
  ((v16h*)panel)[tile * 32 + lane] = o;
}

__global__ void pad_bias(const float* __restrict__ b1, const float* __restrict__ b2,
                         int gateSize, int gateSizePad, int nGates, float* __restrict__ o)
{
  int i = blockIdx.x * blockDim.x + threadIdx.x;
  if (i >= gateSizePad * nGates) return;
  int g = i / gateSizePad, j = i % gateSizePad;
  float v = 0.f;
  if (j < gateSize) { v = b1[g * gateSize + j]; if (b2) v += b2[g * gateSize + j]; }
  o[i] = v;
}

// ---------------- generic WMMA GEMM (optional row-gather A) ----------------
// Register-blocked: each wave computes 4 adjacent 16x16 C tiles along N, sharing one
// A fragment (amortizes the f32->f16 conversion and A loads 4x). 4 waves/block.
// Ntiles4 must be a multiple of 4 (panel is padded); waves are fully in- or out-of-range
// so EXEC stays wave-uniform around the WMMAs.
__global__ void wmma_gemm(const float* __restrict__ A, const int* __restrict__ rowIdx,
                          int aStride, int numRowsA,
                          const _Float16* __restrict__ Bpanel, int Ktiles, int Ntiles4,
                          const float* __restrict__ bias,
                          float* __restrict__ C, int cStride, int cVRows, int cVCols, int act)
{
  int wave = threadIdx.x >> 5;
  int lane = threadIdx.x & 31;
  int nt0 = (blockIdx.x * 4 + wave) * 4;      // base N tile for this wave
  if (nt0 >= Ntiles4) return;                 // wave-uniform exit
  int rtile = blockIdx.y;
  int half = lane >> 4;
  int m = lane & 15;
  int grow = rtile * 16 + m;
  long long rbase;
  if (rowIdx) {
    int rr = (grow < numRowsA) ? rowIdx[grow] : 0;
    rbase = (long long)rr * aStride;
  } else {
    rbase = (grow < numRowsA) ? (long long)grow * aStride : 0;
  }
  const v16h* bp = (const v16h*)Bpanel + (long long)nt0 * Ktiles * 32 + lane;
  v8f acc0 = {0.f,0.f,0.f,0.f,0.f,0.f,0.f,0.f};
  v8f acc1 = acc0, acc2 = acc0, acc3 = acc0;
  for (int kt = 0; kt < Ktiles; ++kt) {
    const float* ap = A + rbase + kt * 32 + half * 8;
    v16h a;
    #pragma unroll
    for (int v = 0; v < 8; ++v) {
      int ko = (v < 4) ? (2 * v) : (16 + 2 * (v - 4));   // ISA 16-bit A-matrix 16x32 layout
      a[2*v]   = (_Float16)ap[ko];
      a[2*v+1] = (_Float16)ap[ko + 1];
    }
    v16h b0 = bp[(long long)(0 * Ktiles + kt) * 32];
    v16h b1 = bp[(long long)(1 * Ktiles + kt) * 32];
    v16h b2 = bp[(long long)(2 * Ktiles + kt) * 32];
    v16h b3 = bp[(long long)(3 * Ktiles + kt) * 32];
    acc0 = __builtin_amdgcn_wmma_f32_16x16x32_f16(false, a, false, b0, (short)0, acc0, false, false);
    acc1 = __builtin_amdgcn_wmma_f32_16x16x32_f16(false, a, false, b1, (short)0, acc1, false, false);
    acc2 = __builtin_amdgcn_wmma_f32_16x16x32_f16(false, a, false, b2, (short)0, acc2, false, false);
    acc3 = __builtin_amdgcn_wmma_f32_16x16x32_f16(false, a, false, b3, (short)0, acc3, false, false);
  }
  #pragma unroll
  for (int s = 0; s < 4; ++s) {
    v8f acc = (s == 0) ? acc0 : (s == 1) ? acc1 : (s == 2) ? acc2 : acc3;
    int col = (nt0 + s) * 16 + m;
    float bv = (bias && col < cVCols) ? bias[col] : 0.f;
    #pragma unroll
    for (int v = 0; v < 8; ++v) {
      int row = rtile * 16 + v + 8 * half;    // C layout: M = vgpr + 8*(lane/16)
      if (row < cVRows && col < cVCols) {
        float val = acc[v] + bv;
        if (act) val = fmaxf(val, 0.f);
        C[(long long)row * cStride + col] = val;
      }
    }
  }
}

// ---------------- LSTM cell elementwise (one timestep) ----------------
__global__ void lstm_cell(const float* __restrict__ gi, const float* __restrict__ gg,
                          float* __restrict__ h, float* __restrict__ c, int t)
{
  int idx = blockIdx.x * blockDim.x + threadIdx.x;
  if (idx >= BN * H_) return;
  int seq = idx / H_, j = idx % H_;
  const float* gr = gi + (long long)(seq * T_STEPS + t) * G4P;
  const float* g2 = gg + (long long)seq * G4P;
  float ig = gr[j]       + g2[j];
  float fg = gr[HP + j]  + g2[HP + j];
  float cg = gr[2*HP + j]+ g2[2*HP + j];
  float og = gr[3*HP + j]+ g2[3*HP + j];
  float cc = sigf(fg) * c[seq * HP + j] + sigf(ig) * tanhf(cg);
  c[seq * HP + j] = cc;
  h[seq * HP + j] = sigf(og) * tanhf(cc);
}

// ---------------- concat [x1(:,0:300), x2(:,0:300)] -> out (stride 608) ----------------
__global__ void concat2(const float* __restrict__ x1, const float* __restrict__ x2,
                        float* __restrict__ out)
{
  int idx = blockIdx.x * blockDim.x + threadIdx.x;
  if (idx >= BN * A_) return;
  int s = idx / A_, j = idx % A_;
  out[(long long)s * AP + j] = (j < H_) ? x1[s * HP + j] : x2[s * HP + (j - H_)];
}

// ---------------- GRU gates elementwise (chunk) ----------------
__global__ void gru_elem(const float* __restrict__ gi, const float* __restrict__ gh,
                         const float* __restrict__ hsl, const int* __restrict__ hidx,
                         float* __restrict__ hprime, int r0, int nrows)
{
  long long idx = (long long)blockIdx.x * blockDim.x + threadIdx.x;
  if (idx >= (long long)nrows * A_) return;
  int row = (int)(idx / A_), j = (int)(idx % A_);
  const float* a = gi + (long long)row * G3P;
  const float* b = gh + (long long)row * G3P;
  float r = sigf(a[j] + b[j]);
  float z = sigf(a[AP + j] + b[AP + j]);
  float n = tanhf(a[2*AP + j] + r * b[2*AP + j]);
  float h = hsl[(long long)hidx[r0 + row] * AP + j];
  hprime[(long long)(r0 + row) * AP + j] = (1.f - z) * n + z * h;
}

// ---------------- sum over axis=1 + soft-norm, accumulate into hsl ----------------
__global__ void reduce_norm(const float* __restrict__ y, float* __restrict__ hsl)
{
  int bk = blockIdx.x;            // 0..255 = b*64+k
  int b = bk >> 6, k = bk & 63;
  int tid = threadIdx.x;          // 256
  float chv[3];
  float n2p = 0.f;
  #pragma unroll
  for (int s = 0; s < 3; ++s) {
    int j = tid + s * 256;
    float acc = 0.f;
    if (j < A_) {
      const float* yp = y + ((long long)(b * 64) * 64 + k) * AP + j;
      for (int i = 0; i < 64; ++i) acc += yp[(long long)i * 64 * AP];
    }
    chv[s] = acc; n2p += acc * acc;
  }
  __shared__ float red[256];
  red[tid] = n2p; __syncthreads();
  for (int s = 128; s > 0; s >>= 1) { if (tid < s) red[tid] += red[tid + s]; __syncthreads(); }
  float n2 = red[0];
  float scale = (n2 + 0.5f) / (n2 + 1.f);
  #pragma unroll
  for (int s = 0; s < 3; ++s) {
    int j = tid + s * 256;
    if (j < A_) hsl[(long long)(bk + 1) * AP + j] += chv[s] * scale;
  }
}

// ---------------- graph construction ----------------
__global__ void scatter_conv(const int* __restrict__ rel, int* __restrict__ sc) {
  int i = blockIdx.x * blockDim.x + threadIdx.x;
  if (i >= 4 * 128) return;
  int b = i >> 7, e = i & 127;
  int a0 = rel[(b * 128 + e) * 2 + 0];
  int a1 = rel[(b * 128 + e) * 2 + 1];
  sc[(b * 64 + a1) * 64 + a0] = 1;
}

__global__ void diag_kernel(const int* __restrict__ idx0, int* __restrict__ dg) {
  int b = blockIdx.x;
  if (threadIdx.x != 0) return;
  int pads = 0;
  for (int n = 0; n < 64; ++n) {
    int v = idx0[(b * 64 + n) * 32 + 0];
    dg[b * 64 + n] = (pads == 0) ? (b * 64 + n + 1) : 0;   // valid = no pad strictly before n
    if (v == 3) pads++;
  }
}

__global__ void build_idx(const int* __restrict__ sc, const int* __restrict__ dg,
                          int* __restrict__ child, int* __restrict__ parent)
{
  int i = blockIdx.x * blockDim.x + threadIdx.x;
  if (i >= BNN) return;
  int b = i >> 12, rem = i & 4095, r = rem >> 6, cj = rem & 63;
  int s = sc[i];
  child[i]  = s ? dg[b * 64 + r]  : 0;   // diag(i) * conv(i,j)
  parent[i] = s ? dg[b * 64 + cj] : 0;   // conv(i,j) * diag(j)
}

// ---------------- host ----------------
static void gemm(hipStream_t st, const float* A, const int* rowIdx, int aStride, int numRowsA,
                 const _Float16* Bp, int Ktiles, int Ntiles4, const float* bias,
                 float* C, int cStride, int cVRows, int cVCols, int act, int rowTiles)
{
  dim3 g((Ntiles4 + 15) / 16, rowTiles);   // 4 waves/block x 4 ntiles/wave
  wmma_gemm<<<g, 128, 0, st>>>(A, rowIdx, aStride, numRowsA, Bp, Ktiles, Ntiles4,
                               bias, C, cStride, cVRows, cVCols, act);
}
static void zero(hipStream_t st, void* p, long long n) {
  fill_zero<<<(unsigned)((n + 255) / 256), 256, 0, st>>>((float*)p, n);
}

extern "C" void kernel_launch(void* const* d_in, const int* in_sizes, int n_in,
                              void* d_out, int out_size, void* d_ws, size_t ws_size,
                              hipStream_t stream)
{
  const float* x     = (const float*)d_in[0];
  const int*   relat = (const int*)d_in[2];
  const int*   idx0  = (const int*)d_in[3];
  const float* wihf = (const float*)d_in[4];  const float* whhf = (const float*)d_in[5];
  const float* bihf = (const float*)d_in[6];  const float* bhhf = (const float*)d_in[7];
  const float* wihb = (const float*)d_in[8];  const float* whhb = (const float*)d_in[9];
  const float* bihb = (const float*)d_in[10]; const float* bhhb = (const float*)d_in[11];
  const float* fhw  = (const float*)d_in[12]; const float* fhb  = (const float*)d_in[13];
  const float* fcw  = (const float*)d_in[14]; const float* fcb  = (const float*)d_in[15];
  const float* fhidw= (const float*)d_in[16]; const float* fhidb= (const float*)d_in[17];
  const float* gfwih= (const float*)d_in[18]; const float* gfwhh= (const float*)d_in[19];
  const float* gfbih= (const float*)d_in[20]; const float* gfbhh= (const float*)d_in[21];
  const float* gfow = (const float*)d_in[22]; const float* gfob = (const float*)d_in[23];
  const float* gbwih= (const float*)d_in[24]; const float* gbwhh= (const float*)d_in[25];
  const float* gbbih= (const float*)d_in[26]; const float* gbbhh= (const float*)d_in[27];
  const float* gbow = (const float*)d_in[28]; const float* gbob = (const float*)d_in[29];
  float* out = (float*)d_out;

  char* ws = (char*)d_ws; size_t off = 0;
  auto alloc = [&](size_t bytes) -> void* {
    off = (off + 255) & ~(size_t)255; void* p = ws + off; off += bytes; return p;
  };
  // f16 weight panels (N-tile counts padded to multiples of 4 for the 4-wide waves)
  _Float16* p_wih_f = (_Float16*)alloc((size_t)80*4*512*2);
  _Float16* p_wih_b = (_Float16*)alloc((size_t)80*4*512*2);
  _Float16* p_whh_f = (_Float16*)alloc((size_t)80*10*512*2);
  _Float16* p_whh_b = (_Float16*)alloc((size_t)80*10*512*2);
  _Float16* p_fh    = (_Float16*)alloc((size_t)20*19*512*2);
  _Float16* p_fc    = (_Float16*)alloc((size_t)20*19*512*2);
  _Float16* p_fhid  = (_Float16*)alloc((size_t)20*19*512*2);
  _Float16* p_gf_ih = (_Float16*)alloc((size_t)116*19*512*2);
  _Float16* p_gf_hh = (_Float16*)alloc((size_t)116*19*512*2);
  _Float16* p_gb_ih = (_Float16*)alloc((size_t)116*19*512*2);
  _Float16* p_gb_hh = (_Float16*)alloc((size_t)116*19*512*2);
  _Float16* p_gf_ow = (_Float16*)alloc((size_t)40*19*512*2);
  _Float16* p_gb_ow = (_Float16*)alloc((size_t)40*19*512*2);
  // padded biases
  float* b_lstm_f = (float*)alloc(G4P*4); float* b_lstm_b = (float*)alloc(G4P*4);
  float* b_fh = (float*)alloc(HP*4); float* b_fc = (float*)alloc(HP*4); float* b_fhid = (float*)alloc(HP*4);
  float* b_gf_ih = (float*)alloc(G3P*4); float* b_gf_hh = (float*)alloc(G3P*4);
  float* b_gb_ih = (float*)alloc(G3P*4); float* b_gb_hh = (float*)alloc(G3P*4);
  float* b_gf_o = (float*)alloc(AP*4);   float* b_gb_o = (float*)alloc(AP*4);
  // big buffers
  float* gi_f = (float*)alloc((size_t)BNT*G4P*4);
  float* gi_b = (float*)alloc((size_t)BNT*G4P*4);
  float* gg   = (float*)alloc((size_t)BN*G4P*4);
  float* h_f = (float*)alloc((size_t)BN*HP*4); float* c_f = (float*)alloc((size_t)BN*HP*4);
  float* h_b = (float*)alloc((size_t)BN*HP*4); float* c_b = (float*)alloc((size_t)BN*HP*4);
  float* ccat = (float*)alloc((size_t)BN*AP*4);
  float* enc_h = (float*)alloc((size_t)BN*HP*4); float* enc_c = (float*)alloc((size_t)BN*HP*4);
  float* hsl = (float*)alloc((size_t)272*AP*4);
  int* sc = (int*)alloc((size_t)BNN*4); int* dgb = (int*)alloc(256*4);
  int* childI = (int*)alloc((size_t)BNN*4); int* parentI = (int*)alloc((size_t)BNN*4);
  float* gi_c = (float*)alloc((size_t)CHUNK*G3P*4);
  float* gh_c = (float*)alloc((size_t)CHUNK*G3P*4);
  float* hprime = gi_f;   // overlay: LSTM gate buffers retired before rounds
  float* y      = gi_b;

  // ---- zero state / padded buffers ----
  zero(stream, h_f, (long long)BN*HP); zero(stream, c_f, (long long)BN*HP);
  zero(stream, h_b, (long long)BN*HP); zero(stream, c_b, (long long)BN*HP);
  zero(stream, ccat, (long long)BN*AP);
  zero(stream, hsl, (long long)272*AP);
  zero(stream, sc, BNN);

  // ---- repack weights + biases ----
  auto rpk = [&](const float* W, int Nout, int K, int gs, int gsp, int Nt, int Kt, _Float16* pan) {
    long long tot = (long long)Nt*Kt*32;
    repack_w<<<(unsigned)((tot+127)/128), 128, 0, stream>>>(W, Nout, K, gs, gsp, Nt, Kt, pan);
  };
  rpk(wihf, 1200, 128, 300, 320, 80, 4,  p_wih_f);
  rpk(wihb, 1200, 128, 300, 320, 80, 4,  p_wih_b);
  rpk(whhf, 1200, 300, 300, 320, 80, 10, p_whh_f);
  rpk(whhb, 1200, 300, 300, 320, 80, 10, p_whh_b);
  rpk(fhw,  300, 600, 300, 320, 20, 19,  p_fh);
  rpk(fcw,  300, 600, 300, 320, 20, 19,  p_fc);
  rpk(fhidw,300, 600, 300, 320, 20, 19,  p_fhid);
  rpk(gfwih,1800, 600, 600, 608, 116, 19, p_gf_ih);
  rpk(gfwhh,1800, 600, 600, 608, 116, 19, p_gf_hh);
  rpk(gbwih,1800, 600, 600, 608, 116, 19, p_gb_ih);
  rpk(gbwhh,1800, 600, 600, 608, 116, 19, p_gb_hh);
  rpk(gfow,  600, 600, 600, 608, 40, 19, p_gf_ow);
  rpk(gbow,  600, 600, 600, 608, 40, 19, p_gb_ow);
  auto pb = [&](const float* b1, const float* b2, int gs, int gsp, int ng, float* o) {
    pad_bias<<<(gsp*ng+255)/256, 256, 0, stream>>>(b1, b2, gs, gsp, ng, o);
  };
  pb(bihf, bhhf, 300, 320, 4, b_lstm_f);
  pb(bihb, bhhb, 300, 320, 4, b_lstm_b);
  pb(fhb, nullptr, 300, 320, 1, b_fh);
  pb(fcb, nullptr, 300, 320, 1, b_fc);
  pb(fhidb, nullptr, 300, 320, 1, b_fhid);
  pb(gfbih, nullptr, 600, 608, 3, b_gf_ih);
  pb(gfbhh, nullptr, 600, 608, 3, b_gf_hh);
  pb(gbbih, nullptr, 600, 608, 3, b_gb_ih);
  pb(gbbhh, nullptr, 600, 608, 3, b_gb_hh);
  pb(gfob, nullptr, 600, 608, 1, b_gf_o);
  pb(gbob, nullptr, 600, 608, 1, b_gb_o);

  // ---- phase 1: time-parallel input GEMMs: gi = x @ W_ih^T + (b_ih + b_hh) ----
  gemm(stream, x, nullptr, D_IN, BNT, p_wih_f, 4, 80, b_lstm_f, gi_f, G4P, BNT, G4P, 0, BNT/16);
  gemm(stream, x, nullptr, D_IN, BNT, p_wih_b, 4, 80, b_lstm_b, gi_b, G4P, BNT, G4P, 0, BNT/16);

  // ---- phase 2: LSTM recurrence ----
  for (int t = 0; t < T_STEPS; ++t) {
    gemm(stream, h_f, nullptr, HP, BN, p_whh_f, 10, 80, nullptr, gg, G4P, BN, G4P, 0, BN/16);
    lstm_cell<<<(BN*H_+255)/256, 256, 0, stream>>>(gi_f, gg, h_f, c_f, t);
  }
  for (int s = 0; s < T_STEPS; ++s) {
    gemm(stream, h_b, nullptr, HP, BN, p_whh_b, 10, 80, nullptr, gg, G4P, BN, G4P, 0, BN/16);
    lstm_cell<<<(BN*H_+255)/256, 256, 0, stream>>>(gi_b, gg, h_b, c_b, T_STEPS - 1 - s);
  }

  // ---- phase 3: encoders + hsl ----
  concat2<<<(BN*A_+255)/256, 256, 0, stream>>>(h_f, c_f, ccat);
  gemm(stream, ccat, nullptr, AP, BN, p_fh, 19, 20, b_fh, enc_h, HP, BN, H_, 1, BN/16);
  concat2<<<(BN*A_+255)/256, 256, 0, stream>>>(h_b, c_b, ccat);
  gemm(stream, ccat, nullptr, AP, BN, p_fc, 19, 20, b_fc, enc_c, HP, BN, H_, 1, BN/16);
  concat2<<<(BN*A_+255)/256, 256, 0, stream>>>(enc_c, enc_h, hsl + AP);  // rows 1..256

  // ---- phase 4: graph indices ----
  scatter_conv<<<2, 256, 0, stream>>>(relat, sc);
  diag_kernel<<<4, 32, 0, stream>>>(idx0, dgb);
  build_idx<<<(BNN+255)/256, 256, 0, stream>>>(sc, dgb, childI, parentI);

  // ---- phase 5: GRU message rounds ----
  zero(stream, hprime, (long long)BNN*AP);
  auto round = [&](const int* xidx, const int* hidx, const _Float16* pih, const _Float16* phh,
                   const float* bih, const float* bhh, const _Float16* pow_, const float* bo) {
    for (int c0 = 0; c0 < BNN; c0 += CHUNK) {
      gemm(stream, hsl, xidx + c0, AP, CHUNK, pih, 19, 116, bih, gi_c, G3P, CHUNK, G3P, 0, CHUNK/16);
      gemm(stream, hsl, hidx + c0, AP, CHUNK, phh, 19, 116, bhh, gh_c, G3P, CHUNK, G3P, 0, CHUNK/16);
      gru_elem<<<(unsigned)(((long long)CHUNK*A_+255)/256), 256, 0, stream>>>(
          gi_c, gh_c, hsl, hidx, hprime, c0, CHUNK);
    }
    gemm(stream, hprime, nullptr, AP, BNN, pow_, 19, 40, bo, y, AP, BNN, AP, 0, BNN/16);
    reduce_norm<<<256, 256, 0, stream>>>(y, hsl);
  };
  round(childI, parentI, p_gb_ih, p_gb_hh, b_gb_ih, b_gb_hh, p_gb_ow, b_gb_o);   // grub first
  round(parentI, childI, p_gf_ih, p_gf_hh, b_gf_ih, b_gf_hh, p_gf_ow, b_gf_o);   // then gruf

  // ---- phase 6: hidden = relu(hsl @ full_hidden_w^T + b) -> (257,300) ----
  gemm(stream, hsl, nullptr, AP, 272, p_fhid, 19, 20, b_fhid, out, H_, 257, H_, 1, 17);

  (void)in_sizes; (void)n_in; (void)out_size; (void)ws_size;
}
